// GPSConv_17549236371615
// MI455X (gfx1250) — matrix operand order repaired and checked by reading, not compiled, not run on hardware
//
#include <hip/hip_runtime.h>

// ---------------------------------------------------------------------------
// GraphGPS block for MI455X (gfx1250, wave32, WMMA + async-to-LDS pipeline).
// GEMMs: v_wmma_f32_16x16x32_bf16, double-buffered LDS tiles filled with
// global_load_async_to_lds_b128 (ASYNCcnt), weights pre-transposed to K-major.
// ---------------------------------------------------------------------------

typedef __bf16 bf16;
typedef __attribute__((ext_vector_type(16))) __bf16 v16bf;
typedef __attribute__((ext_vector_type(8)))  __bf16 v8bf;
typedef __attribute__((ext_vector_type(8)))  float  v8f;

static constexpr int Bb = 32, Tt = 512, Cc = 256, Hh = 8, Dh = 32;
static constexpr int NT = Bb * Tt;          // 16384 dense rows
static constexpr float EPSc = 1e-5f;

#define CDIV(a, b) (((a) + (b) - 1) / (b))

__device__ __forceinline__ v8f wmma_bf16(v16bf a, v16bf b, v8f c) {
  // (neg_a, A, neg_b, B, c_mod, C, reuse_a, reuse_b)
  return __builtin_amdgcn_wmma_f32_16x16x32_bf16(false, a, false, b, (short)0, c,
                                                 false, false);
}

// LDS byte offset = low 32 bits of the generic address (LDS aperture keeps the
// offset in addr[31:0] per the CDNA5 aperture rules).
__device__ __forceinline__ unsigned lds_lo(const void* p) {
  return (unsigned)(uintptr_t)p;
}

// Async DMA of 16 bytes global -> LDS (per-lane addresses, ASYNCcnt-tracked).
__device__ __forceinline__ void async_ld_b128(unsigned ldsoff, const void* g) {
  asm volatile("global_load_async_to_lds_b128 %0, %1, off"
               :: "v"(ldsoff), "v"((unsigned long long)(uintptr_t)g)
               : "memory");
}

template <int N>
__device__ __forceinline__ void wait_async() {
  asm volatile("s_wait_asynccnt %0" :: "i"(N) : "memory");
}

// ----------------------------- small utility kernels -----------------------

__global__ void k_cast(const float* __restrict__ s, bf16* __restrict__ d, long n) {
  long gid = (long)blockIdx.x * blockDim.x + threadIdx.x;
  if (gid < n) d[gid] = (bf16)s[gid];
}

// W (K x M) f32 -> WT (M x K) bf16  (K-major weights for async B-tile staging)
__global__ void k_cast_t(const float* __restrict__ w, bf16* __restrict__ wt,
                         int K, int M) {
  long gid = (long)blockIdx.x * blockDim.x + threadIdx.x;
  if (gid >= (long)K * M) return;
  int k = (int)(gid / M);
  int m = (int)(gid % M);
  wt[(long)m * K + k] = (bf16)w[gid];
}

__global__ void k_zero_f32(float* __restrict__ p, long n) {
  long gid = (long)blockIdx.x * blockDim.x + threadIdx.x;
  if (gid < n) p[gid] = 0.0f;
}

__global__ void k_zero_b16(bf16* __restrict__ p, long n) {
  long gid = (long)blockIdx.x * blockDim.x + threadIdx.x;
  if (gid < n) p[gid] = (bf16)0.0f;
}

// scatter ragged nodes into dense [B*T, C] bf16 (to_dense_batch)
__global__ void k_scatter(const float* __restrict__ x, const int* __restrict__ idx,
                          bf16* __restrict__ xd, int N) {
  long gid = (long)blockIdx.x * blockDim.x + threadIdx.x;
  long total = (long)N * (Cc / 8);
  if (gid >= total) return;
  int i = (int)(gid / (Cc / 8));
  int c = (int)(gid % (Cc / 8)) * 8;
  const float4* s = reinterpret_cast<const float4*>(x + (long)i * Cc + c);
  float4 f0 = s[0], f1 = s[1];
  alignas(16) bf16 o[8];
  o[0] = (bf16)f0.x; o[1] = (bf16)f0.y; o[2] = (bf16)f0.z; o[3] = (bf16)f0.w;
  o[4] = (bf16)f1.x; o[5] = (bf16)f1.y; o[6] = (bf16)f1.z; o[7] = (bf16)f1.w;
  long drow = idx[i];
  *reinterpret_cast<uint4*>(xd + drow * Cc + c) = *reinterpret_cast<uint4*>(o);
}

// segment_sum over edges: agg[dst] += x[src]
__global__ void k_edge_agg(const float* __restrict__ x, const int* __restrict__ ei,
                           float* __restrict__ agg, int E) {
  long gid = (long)blockIdx.x * blockDim.x + threadIdx.x;
  long total = (long)E * (Cc / 4);
  if (gid >= total) return;
  int e = (int)(gid / (Cc / 4));
  int c = (int)(gid % (Cc / 4)) * 4;
  int src = ei[e];
  int dst = ei[E + e];
  const float4 v = *reinterpret_cast<const float4*>(x + (long)src * Cc + c);
  float* d = agg + (long)dst * Cc + c;
  atomicAdd(d + 0, v.x); atomicAdd(d + 1, v.y);
  atomicAdd(d + 2, v.z); atomicAdd(d + 3, v.w);
}

// h_attn = proj[idx[i]] + x[i]
__global__ void k_gather_res(const float* __restrict__ proj, const int* __restrict__ idx,
                             const float* __restrict__ x, float* __restrict__ h, int N) {
  long gid = (long)blockIdx.x * blockDim.x + threadIdx.x;
  long total = (long)N * (Cc / 4);
  if (gid >= total) return;
  int i = (int)(gid / (Cc / 4));
  int c = (int)(gid % (Cc / 4)) * 4;
  long srow = idx[i];
  float4 p = *reinterpret_cast<const float4*>(proj + srow * Cc + c);
  float4 xv = *reinterpret_cast<const float4*>(x + (long)i * Cc + c);
  float4 r; r.x = p.x + xv.x; r.y = p.y + xv.y; r.z = p.z + xv.z; r.w = p.w + xv.w;
  *reinterpret_cast<float4*>(h + (long)i * Cc + c) = r;
}

__global__ void k_combine(const float* __restrict__ a, const float* __restrict__ b,
                          float* __restrict__ of, bf16* __restrict__ ob, long n) {
  long gid = (long)blockIdx.x * blockDim.x + threadIdx.x;
  if (gid >= n) return;
  float v = a[gid] + b[gid];
  of[gid] = v;
  ob[gid] = (bf16)v;
}

// BatchNorm1d (training stats) in two passes
__global__ void k_bn_stats(const float* __restrict__ h, float* __restrict__ sums,
                           long n, int C2) {
  long gid = (long)blockIdx.x * blockDim.x + threadIdx.x;
  if (gid >= n) return;
  int c = (int)(gid % C2);
  float v = h[gid];
  atomicAdd(&sums[c], v);
  atomicAdd(&sums[C2 + c], v * v);
}

__global__ void k_bn_apply(const float* __restrict__ src, float* __restrict__ dst,
                           const float* __restrict__ sums, const float* __restrict__ g,
                           const float* __restrict__ be, long n, int C2, float inv_rows) {
  long gid = (long)blockIdx.x * blockDim.x + threadIdx.x;
  if (gid >= n) return;
  int c = (int)(gid % C2);
  float mean = sums[c] * inv_rows;
  float var  = sums[C2 + c] * inv_rows - mean * mean;   // biased, like jnp.var
  dst[gid] = g[c] * (src[gid] - mean) * rsqrtf(var + EPSc) + be[c];
}

// ----------------------------- WMMA GEMM -----------------------------------
// out = A(rows x K) @ B(K x M) [+ A2 @ B2] [+ bias] [+ resid] [relu]
// B is supplied TRANSPOSED (M x K, K-major) so both A and B tiles are
// contiguous async b128 copies; LDS B tile is directly in fragment layout.
// 128 threads = 4 waves; block tile 64x64; wave tile 16 rows x 64 cols.
// Double-buffered LDS, filled by global_load_async_to_lds_b128.
template <bool DUAL>
__global__ __launch_bounds__(128) void k_gemm(
    const bf16* __restrict__ A, const bf16* __restrict__ BT,
    const bf16* __restrict__ A2, const bf16* __restrict__ BT2,
    const float* __restrict__ bias, const float* __restrict__ resid,
    float* __restrict__ outF, bf16* __restrict__ outB,
    int rows, int K, int M, int relu) {
  __shared__ alignas(64) bf16 At [2][64][32];
  __shared__ alignas(64) bf16 Bt [2][64][32];               // [col][k]
  __shared__ alignas(64) bf16 At2[DUAL ? 2 : 1][64][32];
  __shared__ alignas(64) bf16 Bt2[DUAL ? 2 : 1][64][32];

  const int tid  = threadIdx.x;
  const int lane = tid & 31;
  const int wv   = tid >> 5;     // wave id 0..3
  const int lh   = lane >> 4;    // lane half (ISA fragment layout)
  const int l16  = lane & 15;
  const int row0 = blockIdx.y * 64;
  const int col0 = blockIdx.x * 64;

  v8f acc[4];
#pragma unroll
  for (int j = 0; j < 4; ++j)
#pragma unroll
    for (int r = 0; r < 8; ++r) acc[j][r] = 0.0f;

  const int sr = tid >> 1;            // staged tile row 0..63
  const int sc = (tid & 1) * 16;      // 16-element K segment (32B)
  const int gra = (row0 + sr < rows) ? (row0 + sr) : (rows - 1);  // clamp

  auto stage = [&](int buf, int k0) {
    const bf16* ga = A + (long)gra * K + k0 + sc;
    async_ld_b128(lds_lo(&At[buf][sr][sc]),     ga);
    async_ld_b128(lds_lo(&At[buf][sr][sc + 8]), ga + 8);
    const bf16* gb = BT + (long)(col0 + sr) * K + k0 + sc;
    async_ld_b128(lds_lo(&Bt[buf][sr][sc]),     gb);
    async_ld_b128(lds_lo(&Bt[buf][sr][sc + 8]), gb + 8);
    if constexpr (DUAL) {
      const bf16* ga2 = A2 + (long)gra * K + k0 + sc;
      async_ld_b128(lds_lo(&At2[buf][sr][sc]),     ga2);
      async_ld_b128(lds_lo(&At2[buf][sr][sc + 8]), ga2 + 8);
      const bf16* gb2 = BT2 + (long)(col0 + sr) * K + k0 + sc;
      async_ld_b128(lds_lo(&Bt2[buf][sr][sc]),     gb2);
      async_ld_b128(lds_lo(&Bt2[buf][sr][sc + 8]), gb2 + 8);
    }
  };
  constexpr int OPS = DUAL ? 8 : 4;   // async issues per thread per tile-set

  stage(0, 0);
  const int steps = K >> 5;
  for (int s = 0; s < steps; ++s) {
    const int cur = s & 1;
    if (s + 1 < steps) {
      stage(cur ^ 1, (s + 1) << 5);   // prefetch next tile-set
      wait_async<OPS>();              // in-order: current tile-set has landed
    } else {
      wait_async<0>();
    }
    __syncthreads();

    // fragments in contiguous k order (k = 16*lh + s), same for A and B
    const int am = wv * 16 + l16;
    const int kh = lh * 8;
    const int kb16 = lh * 16;
    v16bf a;
    *reinterpret_cast<v8bf*>(&a)       = *reinterpret_cast<const v8bf*>(&At[cur][am][kb16]);
    *(reinterpret_cast<v8bf*>(&a) + 1) = *reinterpret_cast<const v8bf*>(&At[cur][am][kb16 + 8]);
    v16bf bfr[4];
#pragma unroll
    for (int j = 0; j < 4; ++j)
      bfr[j] = *reinterpret_cast<const v16bf*>(&Bt[cur][j * 16 + l16][kb16]);
#pragma unroll
    for (int j = 0; j < 4; ++j) acc[j] = wmma_bf16(a, bfr[j], acc[j]);
    if constexpr (DUAL) {
      v16bf a2;
      *reinterpret_cast<v8bf*>(&a2)       = *reinterpret_cast<const v8bf*>(&At2[cur][am][kb16]);
      *(reinterpret_cast<v8bf*>(&a2) + 1) = *reinterpret_cast<const v8bf*>(&At2[cur][am][kb16 + 8]);
      v16bf b2r[4];
#pragma unroll
      for (int j = 0; j < 4; ++j)
        b2r[j] = *reinterpret_cast<const v16bf*>(&Bt2[cur][j * 16 + l16][kb16]);
#pragma unroll
      for (int j = 0; j < 4; ++j) acc[j] = wmma_bf16(a2, b2r[j], acc[j]);
    }
    __syncthreads();   // all waves done reading before next overwrite
  }

  // epilogue: C/D layout lane(l16)=col, VGPR r = row (lh*8 + r)
#pragma unroll
  for (int j = 0; j < 4; ++j) {
    const int gc = col0 + j * 16 + l16;
    const float bs = bias ? bias[gc] : 0.0f;
#pragma unroll
    for (int r = 0; r < 8; ++r) {
      const int gr = row0 + wv * 16 + lh * 8 + r;
      if (gr < rows) {
        float v = acc[j][r] + bs;
        if (resid) v += resid[(long)gr * M + gc];
        if (relu) v = fmaxf(v, 0.0f);
        if (outF) outF[(long)gr * M + gc] = v;
        if (outB) outB[(long)gr * M + gc] = (bf16)v;
      }
    }
  }
}

// ----------------------------- flash-style MHA ------------------------------
// One wave per (batch, head, 16-query tile). Dh=32 -> one WMMA per score tile.
__global__ __launch_bounds__(32) void k_attn(
    const bf16* __restrict__ qm, const bf16* __restrict__ km,
    const bf16* __restrict__ vm, const unsigned char* __restrict__ maskp,
    bf16* __restrict__ om) {
  __shared__ alignas(64) bf16 pbuf[16][32];   // exp(scores) relayout C->A
  __shared__ alignas(64) bf16 vt[32][32];     // V tile transposed [dh][key]

  const int qt = blockIdx.x;   // query tile 0..31
  const int h  = blockIdx.y;
  const int b  = blockIdx.z;
  const int lane = threadIdx.x;
  const int lh = lane >> 4, l16 = lane & 15;

  // Q fragment: A-matrix 16x32 (rows = queries, K = Dh)
  v16bf aq;
  {
    const bf16* qp = qm + (long)(b * Tt + qt * 16 + l16) * Cc + h * Dh;
    *reinterpret_cast<v8bf*>(&aq)       = *reinterpret_cast<const v8bf*>(qp + lh * 16);
    *(reinterpret_cast<v8bf*>(&aq) + 1) = *reinterpret_cast<const v8bf*>(qp + lh * 16 + 8);
  }

  float mrun[8], lrun[8];
  v8f accO[2];
#pragma unroll
  for (int r = 0; r < 8; ++r) {
    mrun[r] = -3.0e38f; lrun[r] = 0.0f; accO[0][r] = 0.0f; accO[1][r] = 0.0f;
  }
  const float scale = 0.1767766952966369f;   // 1/sqrt(32)

  for (int kb = 0; kb < Tt; kb += 32) {
    // stage V tile transposed: lane = key index
    {
      const bf16* vr = vm + (long)(b * Tt + kb + lane) * Cc + h * Dh;
      uint4 u[4];
      const uint4* s = reinterpret_cast<const uint4*>(vr);
      u[0] = s[0]; u[1] = s[1]; u[2] = s[2]; u[3] = s[3];
      const bf16* e = reinterpret_cast<const bf16*>(u);
#pragma unroll
      for (int d = 0; d < 32; ++d) vt[d][lane] = e[d];
    }

    // two 16x16 score tiles: B-fragment = K^T, contiguous along Dh
    v8f s0, s1;
    {
      v8f z;
#pragma unroll
      for (int r = 0; r < 8; ++r) z[r] = 0.0f;
      const bf16* kp0 = km + (long)(b * Tt + kb + l16) * Cc + h * Dh + lh * 16;
      const bf16* kp1 = km + (long)(b * Tt + kb + 16 + l16) * Cc + h * Dh + lh * 16;
      v16bf bk0 = *reinterpret_cast<const v16bf*>(kp0);
      v16bf bk1 = *reinterpret_cast<const v16bf*>(kp1);
      s0 = wmma_bf16(aq, bk0, z);
      s1 = wmma_bf16(aq, bk1, z);
    }
    const bool ok0 = maskp[b * Tt + kb + l16] != 0;
    const bool ok1 = maskp[b * Tt + kb + 16 + l16] != 0;
    __syncthreads();   // vt staged

    // online softmax update; row-wise reductions across 16-lane groups
#pragma unroll
    for (int r = 0; r < 8; ++r) {
      float v0 = ok0 ? s0[r] * scale : -1e9f;
      float v1 = ok1 ? s1[r] * scale : -1e9f;
      float cur = fmaxf(v0, v1);
#pragma unroll
      for (int off = 1; off < 16; off <<= 1) cur = fmaxf(cur, __shfl_xor(cur, off, 16));
      const float mn = fmaxf(mrun[r], cur);
      const float p0 = __expf(v0 - mn);
      const float p1 = __expf(v1 - mn);
      float rs = p0 + p1;
#pragma unroll
      for (int off = 1; off < 16; off <<= 1) rs += __shfl_xor(rs, off, 16);
      const float corr = __expf(mrun[r] - mn);
      lrun[r] = lrun[r] * corr + rs;
      accO[0][r] *= corr;
      accO[1][r] *= corr;
      mrun[r] = mn;
      pbuf[lh * 8 + r][l16]      = (bf16)p0;   // C-layout -> LDS
      pbuf[lh * 8 + r][16 + l16] = (bf16)p1;
    }
    __syncthreads();   // pbuf written

    // P (16x32) A-fragment and V (32x16) B-fragments -> accumulate output
    v16bf ap;
    *reinterpret_cast<v8bf*>(&ap)       = *reinterpret_cast<const v8bf*>(&pbuf[l16][lh * 16]);
    *(reinterpret_cast<v8bf*>(&ap) + 1) = *reinterpret_cast<const v8bf*>(&pbuf[l16][lh * 16 + 8]);
#pragma unroll
    for (int nt = 0; nt < 2; ++nt) {
      v16bf bv = *reinterpret_cast<const v16bf*>(&vt[nt * 16 + l16][lh * 16]);
      accO[nt] = wmma_bf16(ap, bv, accO[nt]);
    }
    __syncthreads();   // before vt/pbuf reuse
  }

#pragma unroll
  for (int nt = 0; nt < 2; ++nt)
#pragma unroll
    for (int r = 0; r < 8; ++r) {
      const float v = accO[nt][r] / lrun[r];
      om[(long)(b * Tt + qt * 16 + lh * 8 + r) * Cc + h * Dh + nt * 16 + l16] = (bf16)v;
    }
}

// ----------------------------- host orchestration ---------------------------

extern "C" void kernel_launch(void* const* d_in, const int* in_sizes, int n_in,
                              void* d_out, int out_size, void* d_ws, size_t ws_size,
                              hipStream_t stream) {
  (void)n_in; (void)out_size; (void)ws_size;
  const float* x    = (const float*)d_in[0];
  const int*   ei   = (const int*)d_in[1];
  const int*   idx  = (const int*)d_in[2];
  const unsigned char* mask = (const unsigned char*)d_in[3];
  const float* W_root = (const float*)d_in[4];
  const float* W_nbr  = (const float*)d_in[5];
  const float* b_conv = (const float*)d_in[6];
  const float* Wq = (const float*)d_in[7];
  const float* Wk = (const float*)d_in[8];
  const float* Wv = (const float*)d_in[9];
  const float* bq = (const float*)d_in[10];
  const float* bk = (const float*)d_in[11];
  const float* bv = (const float*)d_in[12];
  const float* Wo = (const float*)d_in[13];
  const float* bo = (const float*)d_in[14];
  const float* W1 = (const float*)d_in[15];
  const float* b1 = (const float*)d_in[16];
  const float* W2 = (const float*)d_in[17];
  const float* b2 = (const float*)d_in[18];
  const float* g1 = (const float*)d_in[19];
  const float* be1 = (const float*)d_in[20];
  const float* g2 = (const float*)d_in[21];
  const float* be2 = (const float*)d_in[22];
  const float* g3 = (const float*)d_in[23];
  const float* be3 = (const float*)d_in[24];

  const int N = in_sizes[0] / Cc;
  const int E = in_sizes[1] / 2;

  char* p = (char*)d_ws;
  auto alloc = [&](size_t bytes) -> void* {
    void* r = (void*)p;
    p += (bytes + 255) & ~(size_t)255;
    return r;
  };
  bf16*  xb    = (bf16*) alloc((size_t)N * Cc * 2);
  float* aggf  = (float*)alloc((size_t)N * Cc * 4);
  bf16*  aggb  = (bf16*) alloc((size_t)N * Cc * 2);
  float* hloc  = (float*)alloc((size_t)N * Cc * 4);
  bf16*  xdb   = (bf16*) alloc((size_t)NT * Cc * 2);
  bf16*  qbuf  = (bf16*) alloc((size_t)NT * Cc * 2);
  bf16*  kbuf  = (bf16*) alloc((size_t)NT * Cc * 2);
  bf16*  vbuf  = (bf16*) alloc((size_t)NT * Cc * 2);
  bf16*  obuf  = (bf16*) alloc((size_t)NT * Cc * 2);
  float* projf = (float*)alloc((size_t)NT * Cc * 4);
  float* hattn = (float*)alloc((size_t)N * Cc * 4);
  float* outf  = (float*)alloc((size_t)N * Cc * 4);
  bf16*  outb  = (bf16*) alloc((size_t)N * Cc * 2);
  bf16*  hidb  = (bf16*) alloc((size_t)N * 2 * Cc * 2);
  float* out2  = (float*)alloc((size_t)N * Cc * 4);
  bf16* wtr = (bf16*)alloc((size_t)Cc * Cc * 2);       // transposed bf16 weights
  bf16* wtn = (bf16*)alloc((size_t)Cc * Cc * 2);
  bf16* wtq = (bf16*)alloc((size_t)Cc * Cc * 2);
  bf16* wtk = (bf16*)alloc((size_t)Cc * Cc * 2);
  bf16* wtv = (bf16*)alloc((size_t)Cc * Cc * 2);
  bf16* wto = (bf16*)alloc((size_t)Cc * Cc * 2);
  bf16* wt1 = (bf16*)alloc((size_t)Cc * 2 * Cc * 2);
  bf16* wt2 = (bf16*)alloc((size_t)2 * Cc * Cc * 2);
  float* sums = (float*)alloc((size_t)2 * 2 * Cc * 4);

  const int TB = 256;
  auto gs = [&](long n) { return dim3((unsigned)CDIV(n, (long)TB)); };
  const long NC = (long)N * Cc;

  // weights -> transposed bf16 (K-major)
  k_cast_t<<<gs(Cc * Cc), TB, 0, stream>>>(W_root, wtr, Cc, Cc);
  k_cast_t<<<gs(Cc * Cc), TB, 0, stream>>>(W_nbr,  wtn, Cc, Cc);
  k_cast_t<<<gs(Cc * Cc), TB, 0, stream>>>(Wq, wtq, Cc, Cc);
  k_cast_t<<<gs(Cc * Cc), TB, 0, stream>>>(Wk, wtk, Cc, Cc);
  k_cast_t<<<gs(Cc * Cc), TB, 0, stream>>>(Wv, wtv, Cc, Cc);
  k_cast_t<<<gs(Cc * Cc), TB, 0, stream>>>(Wo, wto, Cc, Cc);
  k_cast_t<<<gs((long)Cc * 2 * Cc), TB, 0, stream>>>(W1, wt1, Cc, 2 * Cc);
  k_cast_t<<<gs((long)2 * Cc * Cc), TB, 0, stream>>>(W2, wt2, 2 * Cc, Cc);

  // activations
  k_cast<<<gs(NC), TB, 0, stream>>>(x, xb, NC);
  k_zero_f32<<<gs(NC), TB, 0, stream>>>(aggf, NC);
  k_zero_b16<<<gs((long)NT * Cc), TB, 0, stream>>>(xdb, (long)NT * Cc);
  k_scatter<<<gs((long)N * (Cc / 8)), TB, 0, stream>>>(x, idx, xdb, N);
  k_edge_agg<<<gs((long)E * (Cc / 4)), TB, 0, stream>>>(x, ei, aggf, E);
  k_cast<<<gs(NC), TB, 0, stream>>>(aggf, aggb, NC);

  const dim3 gblk(128);
  // local branch: x@W_root + agg@W_nbr + b_conv + x (residual), then BN1
  k_gemm<true><<<dim3(Cc / 64, CDIV(N, 64)), gblk, 0, stream>>>(
      xb, wtr, aggb, wtn, b_conv, x, hloc, nullptr, N, Cc, Cc, 0);
  k_zero_f32<<<gs(2 * Cc), TB, 0, stream>>>(sums, 2 * Cc);
  k_bn_stats<<<gs(NC), TB, 0, stream>>>(hloc, sums, NC, Cc);
  k_bn_apply<<<gs(NC), TB, 0, stream>>>(hloc, hloc, sums, g1, be1, NC, Cc, 1.0f / N);

  // global branch: QKV projections, flash attention, output projection
  k_gemm<false><<<dim3(Cc / 64, NT / 64), gblk, 0, stream>>>(
      xdb, wtq, nullptr, nullptr, bq, nullptr, nullptr, qbuf, NT, Cc, Cc, 0);
  k_gemm<false><<<dim3(Cc / 64, NT / 64), gblk, 0, stream>>>(
      xdb, wtk, nullptr, nullptr, bk, nullptr, nullptr, kbuf, NT, Cc, Cc, 0);
  k_gemm<false><<<dim3(Cc / 64, NT / 64), gblk, 0, stream>>>(
      xdb, wtv, nullptr, nullptr, bv, nullptr, nullptr, vbuf, NT, Cc, Cc, 0);
  k_attn<<<dim3(Tt / 16, Hh, Bb), dim3(32), 0, stream>>>(qbuf, kbuf, vbuf, mask, obuf);
  k_gemm<false><<<dim3(Cc / 64, NT / 64), gblk, 0, stream>>>(
      obuf, wto, nullptr, nullptr, bo, nullptr, projf, nullptr, NT, Cc, Cc, 0);
  k_gather_res<<<gs((long)N * (Cc / 4)), TB, 0, stream>>>(projf, idx, x, hattn, N);
  k_zero_f32<<<gs(2 * Cc), TB, 0, stream>>>(sums, 2 * Cc);
  k_bn_stats<<<gs(NC), TB, 0, stream>>>(hattn, sums, NC, Cc);
  k_bn_apply<<<gs(NC), TB, 0, stream>>>(hattn, hattn, sums, g2, be2, NC, Cc, 1.0f / N);

  // combine + MLP (256 -> 512 relu -> 256) with residual, then BN3 -> d_out
  k_combine<<<gs(NC), TB, 0, stream>>>(hloc, hattn, outf, outb, NC);
  k_gemm<false><<<dim3(2 * Cc / 64, CDIV(N, 64)), gblk, 0, stream>>>(
      outb, wt1, nullptr, nullptr, b1, nullptr, nullptr, hidb, N, Cc, 2 * Cc, 1);
  k_gemm<false><<<dim3(Cc / 64, CDIV(N, 64)), gblk, 0, stream>>>(
      hidb, wt2, nullptr, nullptr, b2, outf, out2, nullptr, N, 2 * Cc, Cc, 0);
  k_zero_f32<<<gs(2 * Cc), TB, 0, stream>>>(sums, 2 * Cc);
  k_bn_stats<<<gs(NC), TB, 0, stream>>>(out2, sums, NC, Cc);
  k_bn_apply<<<gs(NC), TB, 0, stream>>>(out2, (float*)d_out, sums, g3, be3, NC, Cc, 1.0f / N);
}
